// RecurrentContentTransfer_52776558133392
// MI455X (gfx1250) — matrix operand order repaired and checked by reading, not compiled, not run on hardware
//
#include <hip/hip_runtime.h>
#include <math.h>

typedef __bf16 bf16_t;
typedef __bf16 v16bf __attribute__((ext_vector_type(16)));
typedef __bf16 v8bf  __attribute__((ext_vector_type(8)));
typedef float  v8f   __attribute__((ext_vector_type(8)));

#define FDIM   2048
#define NBATCH 16
#define G4     (4 * FDIM)          // 8192 gate rows
#define HSZ    (NBATCH * FDIM)     // 32768 state elements

union V16U { v16bf v; v8bf h[2]; };

__device__ __forceinline__ float sigmoidf_(float x) {
  return 1.0f / (1.0f + __expf(-x));
}

// ---------------- fp32 -> bf16 weight conversion ----------------
__global__ void k_cvt_bf16(const float* __restrict__ in, bf16_t* __restrict__ out, int n) {
  int i  = blockIdx.x * blockDim.x + threadIdx.x;
  int st = gridDim.x * blockDim.x;
  for (; i < n; i += st) out[i] = (bf16_t)in[i];
}

__global__ void k_bias_sum(const float* __restrict__ a, const float* __restrict__ b,
                           float* __restrict__ o, int n) {
  int i = blockIdx.x * blockDim.x + threadIdx.x;
  if (i < n) o[i] = a[i] + b[i];
}

__global__ void k_zero(float* c0, float* c1, bf16_t* h0a, bf16_t* h0b,
                       bf16_t* h1a, bf16_t* h1b) {
  int i = blockIdx.x * blockDim.x + threadIdx.x;
  if (i < HSZ) {
    c0[i] = 0.f; c1[i] = 0.f;
    bf16_t z = (bf16_t)0.f;
    h0a[i] = z; h0b[i] = z; h1a[i] = z; h1b[i] = z;
  }
}

// ---------------- bottleneck 1x1 conv: latent (16,256,32,32) -> seq (32,16,2048) bf16 ----
__global__ void k_bottleneck(const float* __restrict__ latent,
                             const float* __restrict__ w_in,
                             const float* __restrict__ b_in,
                             bf16_t* __restrict__ seq) {
  __shared__ float lat[256];
  int bid = blockIdx.x;            // packs (n, w, h)
  int h   = bid & 31;
  int w   = (bid >> 5) & 31;
  int n   = bid >> 10;
  int cb  = threadIdx.x;           // 64 threads
  for (int c = cb; c < 256; c += 64)
    lat[c] = latent[(((n * 256) + c) * 32 + w) * 32 + h];
  __syncthreads();
  const float* wr = w_in + cb * 256;
  float acc = b_in[cb];
#pragma unroll 4
  for (int c = 0; c < 256; ++c) acc = fmaf(lat[c], wr[c], acc);
  seq[(w * NBATCH + n) * FDIM + h * 64 + cb] = (bf16_t)acc;
}

// ---------------- fused LSTM cell: gates = x@w_ih^T + h@w_hh^T + b, then state update ----
// grid = 128 (one 16-wide hidden tile each), block = 256 (8 waves = 4 gates x 2 K-halves)
__launch_bounds__(256)
__global__ void k_lstm_cell(const bf16_t* __restrict__ x,      // (16,2048)
                            const bf16_t* __restrict__ hprev,  // (16,2048)
                            const bf16_t* __restrict__ w_ih,   // (8192,2048)
                            const bf16_t* __restrict__ w_hh,   // (8192,2048)
                            const float*  __restrict__ bias,   // (8192) = b_ih + b_hh
                            float*  __restrict__ cstate,       // (16,2048) in/out
                            bf16_t* __restrict__ hout,         // (16,2048)
                            float*  __restrict__ hout_f32) {   // optional f32 copy
  __shared__ float gsm[8 * 256];
  const int tid   = threadIdx.x;
  const int lane  = tid & 31;
  const int wv    = tid >> 5;        // 0..7
  const int gate  = wv & 3;          // i,f,g,o
  const int khalf = wv >> 2;         // K split: [0,1024) / [1024,2048)
  const int tile  = blockIdx.x;      // hidden cols tile*16 .. tile*16+15

  const int m   = lane & 15;         // A row (batch) and B column (N within tile)
  const int hiK = lane >> 4;         // high-K lane group
  const int kb  = khalf * (FDIM / 2);

  // A fragments: per-lane 8 contiguous bf16 at K+ (hiK?8:0) and 8 more at +16.
  const bf16_t* pax = x     + (size_t)m * FDIM + kb + hiK * 8;
  const bf16_t* pah = hprev + (size_t)m * FDIM + kb + hiK * 8;
  // B fragment: weight row j = gate*F + tile*16 + m, 8 contiguous bf16 at K+(hiK?8:0) and 8 more at +16.
  const size_t jrow = (size_t)(gate * FDIM + tile * 16 + m);
  const bf16_t* pbi = w_ih + jrow * FDIM + kb + hiK * 8;
  const bf16_t* pbh = w_hh + jrow * FDIM + kb + hiK * 8;

  v8f acc = {0.f, 0.f, 0.f, 0.f, 0.f, 0.f, 0.f, 0.f};
#pragma unroll 2
  for (int k = 0; k < FDIM / 2; k += 32) {
    V16U ax, ah, bi, bh;
    ax.h[0] = *(const v8bf*)(pax + k);
    ax.h[1] = *(const v8bf*)(pax + k + 16);
    ah.h[0] = *(const v8bf*)(pah + k);
    ah.h[1] = *(const v8bf*)(pah + k + 16);
    bi.h[0] = *(const v8bf*)(pbi + k);
    bi.h[1] = *(const v8bf*)(pbi + k + 16);
    bh.h[0] = *(const v8bf*)(pbh + k);
    bh.h[1] = *(const v8bf*)(pbh + k + 16);
    acc = __builtin_amdgcn_wmma_f32_16x16x32_bf16(false, ax.v, false, bi.v,
                                                  (short)0, acc, false, false);
    acc = __builtin_amdgcn_wmma_f32_16x16x32_bf16(false, ah.v, false, bh.v,
                                                  (short)0, acc, false, false);
  }

  // C layout: VGPR r, lane -> (M = hiK*8 + r, N = m)
  const int mbase = hiK * 8;
#pragma unroll
  for (int r = 0; r < 8; ++r)
    gsm[wv * 256 + (mbase + r) * 16 + m] = acc[r];
  __syncthreads();

  // per-element state update: 256 threads, one (batch, hidden) element each
  {
    int e   = tid;
    int mm  = e >> 4;
    int nn  = e & 15;
    int col = tile * 16 + nn;
    float gi = gsm[0 * 256 + e] + gsm[4 * 256 + e] + bias[0 * FDIM + col];
    float gf = gsm[1 * 256 + e] + gsm[5 * 256 + e] + bias[1 * FDIM + col];
    float gg = gsm[2 * 256 + e] + gsm[6 * 256 + e] + bias[2 * FDIM + col];
    float go = gsm[3 * 256 + e] + gsm[7 * 256 + e] + bias[3 * FDIM + col];
    int idx  = mm * FDIM + col;
    float cn = sigmoidf_(gf) * cstate[idx] + sigmoidf_(gi) * tanhf(gg);
    float hn = sigmoidf_(go) * tanhf(cn);
    cstate[idx] = cn;
    hout[idx]   = (bf16_t)hn;
    if (hout_f32) hout_f32[idx] = hn;
  }
}

// ---------------- projection: seq_out (32,16,2048) f32 -> out[:, :, 32:64, :] ----------
__global__ void k_project(const float* __restrict__ seq_out,
                          const float* __restrict__ w_out,
                          const float* __restrict__ b_out,
                          float* __restrict__ out) {
  __shared__ float s[64];
  int bid = blockIdx.x;            // packs (n, w2, h)
  int h   = bid & 31;
  int w2  = (bid >> 5) & 31;
  int n   = bid >> 10;
  int o   = threadIdx.x;           // 256 threads
  if (o < 64) s[o] = seq_out[(w2 * NBATCH + n) * FDIM + h * 64 + o];
  __syncthreads();
  const float* wr = w_out + o * 64;
  float acc = b_out[o];
#pragma unroll 4
  for (int cb = 0; cb < 64; ++cb) acc = fmaf(s[cb], wr[cb], acc);
  out[(((n * 256) + o) * 64 + (32 + w2)) * 32 + h] = acc;
}

// ---------------- copy latent into out[:, :, 0:32, :] ----------------
__global__ void k_copy_latent(const float* __restrict__ latent, float* __restrict__ out) {
  int i = blockIdx.x * blockDim.x + threadIdx.x;   // 16*256*32*32 = 4194304
  int h  = i & 31;
  int w  = (i >> 5) & 31;
  int oc = (i >> 10) & 255;
  int n  = i >> 18;
  out[(((n * 256) + oc) * 64 + w) * 32 + h] = latent[i];
}

extern "C" void kernel_launch(void* const* d_in, const int* in_sizes, int n_in,
                              void* d_out, int out_size, void* d_ws, size_t ws_size,
                              hipStream_t stream) {
  (void)in_sizes; (void)n_in; (void)out_size; (void)ws_size;
  const float* latent = (const float*)d_in[0];
  const float* w_in   = (const float*)d_in[1];
  const float* b_in   = (const float*)d_in[2];
  const float* w_ih0  = (const float*)d_in[3];
  const float* w_hh0  = (const float*)d_in[4];
  const float* b_ih0  = (const float*)d_in[5];
  const float* b_hh0  = (const float*)d_in[6];
  const float* w_ih1  = (const float*)d_in[7];
  const float* w_hh1  = (const float*)d_in[8];
  const float* b_ih1  = (const float*)d_in[9];
  const float* b_hh1  = (const float*)d_in[10];
  const float* w_out  = (const float*)d_in[11];
  const float* b_out  = (const float*)d_in[12];
  float* out = (float*)d_out;

  // workspace carve (bump allocator, 256B aligned); total ~141 MB
  unsigned char* p = (unsigned char*)d_ws;
  auto carve = [&](size_t bytes) -> void* {
    void* r = (void*)p;
    p += (bytes + 255) & ~(size_t)255;
    return r;
  };
  const size_t WELEMS = (size_t)G4 * FDIM;           // 16,777,216 per matrix
  bf16_t* wih0b = (bf16_t*)carve(WELEMS * 2);
  bf16_t* whh0b = (bf16_t*)carve(WELEMS * 2);
  bf16_t* wih1b = (bf16_t*)carve(WELEMS * 2);
  bf16_t* whh1b = (bf16_t*)carve(WELEMS * 2);
  float*  b0    = (float*)carve((size_t)G4 * 4);
  float*  b1    = (float*)carve((size_t)G4 * 4);
  bf16_t* seq   = (bf16_t*)carve((size_t)32 * HSZ * 2);
  float*  seqo  = (float*)carve((size_t)32 * HSZ * 4);
  bf16_t* h0buf[2] = {(bf16_t*)carve(HSZ * 2), (bf16_t*)carve(HSZ * 2)};
  bf16_t* h1buf[2] = {(bf16_t*)carve(HSZ * 2), (bf16_t*)carve(HSZ * 2)};
  float*  c0    = (float*)carve((size_t)HSZ * 4);
  float*  c1    = (float*)carve((size_t)HSZ * 4);

  // one-time per launch: weights -> bf16 (L2-resident thereafter), fused biases
  k_cvt_bf16<<<16384, 256, 0, stream>>>(w_ih0, wih0b, (int)WELEMS);
  k_cvt_bf16<<<16384, 256, 0, stream>>>(w_hh0, whh0b, (int)WELEMS);
  k_cvt_bf16<<<16384, 256, 0, stream>>>(w_ih1, wih1b, (int)WELEMS);
  k_cvt_bf16<<<16384, 256, 0, stream>>>(w_hh1, whh1b, (int)WELEMS);
  k_bias_sum<<<(G4 + 255) / 256, 256, 0, stream>>>(b_ih0, b_hh0, b0, G4);
  k_bias_sum<<<(G4 + 255) / 256, 256, 0, stream>>>(b_ih1, b_hh1, b1, G4);
  k_bottleneck<<<16384, 64, 0, stream>>>(latent, w_in, b_in, seq);
  k_zero<<<(HSZ + 255) / 256, 256, 0, stream>>>(c0, c1, h0buf[0], h0buf[1],
                                                h1buf[0], h1buf[1]);

  // 63 recurrent steps: 32 teacher-forced + 31 autoregressive; ping-pong h buffers
  int r0 = 0, r1 = 0;
  for (int t = 0; t < 63; ++t) {
    const bf16_t* xt = (t < 32) ? (seq + (size_t)t * HSZ) : h1buf[r1];
    k_lstm_cell<<<128, 256, 0, stream>>>(xt, h0buf[r0], wih0b, whh0b, b0,
                                         c0, h0buf[1 - r0], nullptr);
    float* of = (t >= 31) ? (seqo + (size_t)(t - 31) * HSZ) : nullptr;
    k_lstm_cell<<<128, 256, 0, stream>>>(h0buf[1 - r0], h1buf[r1], wih1b, whh1b, b1,
                                         c1, h1buf[1 - r1], of);
    r0 = 1 - r0;
    r1 = 1 - r1;
  }

  // assemble output (16,256,64,32): [latent | projected generation]
  k_copy_latent<<<16384, 256, 0, stream>>>(latent, out);
  k_project<<<16384, 256, 0, stream>>>(seqo, w_out, b_out, out);
}
